// FlashInternImage_46127948759518
// MI455X (gfx1250) — compile-verified, matrix operand order checked by reading, not run on hardware
//
#include <hip/hip_runtime.h>
#include <hip/hip_bf16.h>

// ---------------------------------------------------------------------------
// DCNv4 for MI455X (gfx1250, wave32, WMMA)
//   N=8, H=W=56, L=3136, C=256, G=8, gc=32, P=9, K=72
//   proj columns: [0,256) value | [256,400) offset | [400,472) mask | pad->512
// Precision: 3x-bf16 split GEMM (hi*hi + hi*lo + lo*hi, fp32 accum) ~= fp32.
// B-tile staging uses the gfx1250 async global->LDS path when available.
// ---------------------------------------------------------------------------

#define NB    8
#define HH    56
#define WW    56
#define LL    (HH * WW)          // 3136
#define CC    256
#define GG    8
#define GC    32
#define PP    9
#define NCAT  512                // 256 + 144 + 72 padded to 512
#define KDIM  256
#define MTOT  (NB * LL)          // 25088

typedef __attribute__((ext_vector_type(16))) __bf16 v16bf;
typedef __attribute__((ext_vector_type(8)))  __bf16 v8bf;
typedef __attribute__((ext_vector_type(8)))  float  v8f;
typedef int vint4 __attribute__((vector_size(16)));

#if __has_builtin(__builtin_amdgcn_global_load_async_to_lds_b128)
#define HAVE_ASYNC_LDS 1
#endif

__device__ __forceinline__ void split_bf16(float f, __bf16* hi, __bf16* lo) {
    __bf16 h = (__bf16)f;
    *hi = h;
    *lo = (__bf16)(f - (float)h);
}

// ---------------------------------------------------------------------------
// Kernel 1: fp32 -> (hi,lo) bf16 splits + weight/bias concatenation
// ---------------------------------------------------------------------------
__global__ void dcn_convert(const float* __restrict__ x,
                            const float* __restrict__ vw, const float* __restrict__ vb,
                            const float* __restrict__ ow, const float* __restrict__ ob,
                            const float* __restrict__ mw, const float* __restrict__ mb,
                            const float* __restrict__ outw,
                            __bf16* __restrict__ xhi,   __bf16* __restrict__ xlo,
                            __bf16* __restrict__ wchi,  __bf16* __restrict__ wclo,
                            __bf16* __restrict__ wohi,  __bf16* __restrict__ wolo,
                            float*  __restrict__ biascat) {
    const int tid    = blockIdx.x * blockDim.x + threadIdx.x;
    const int stride = gridDim.x * blockDim.x;

    // x: (25088, 256)
    for (size_t i = tid; i < (size_t)MTOT * CC; i += stride)
        split_bf16(x[i], &xhi[i], &xlo[i]);

    // wcat: (512, 256)  [value | offset | mask | zeros]
    for (size_t i = tid; i < (size_t)NCAT * KDIM; i += stride) {
        int n = (int)(i / KDIM);
        int k = (int)(i % KDIM);
        float v = 0.0f;
        if      (n < 256) v = vw[(size_t)n * KDIM + k];
        else if (n < 400) v = ow[(size_t)(n - 256) * KDIM + k];
        else if (n < 472) v = mw[(size_t)(n - 400) * KDIM + k];
        split_bf16(v, &wchi[i], &wclo[i]);
    }

    // output_w: (256, 256)
    for (size_t i = tid; i < (size_t)CC * KDIM; i += stride)
        split_bf16(outw[i], &wohi[i], &wolo[i]);

    // concatenated bias (512)
    for (int n = tid; n < NCAT; n += stride) {
        float v = 0.0f;
        if      (n < 256) v = vb[n];
        else if (n < 400) v = ob[n - 256];
        else if (n < 472) v = mb[n - 400];
        biascat[n] = v;
    }
}

// ---------------------------------------------------------------------------
// 3x-bf16 WMMA GEMM:  C[m,n] = sum_k A[m,k]*B[n,k] + bias[n]   (fp32 accurate)
//   A(hi/lo): (M,K) bf16 row-major, lda=K.  B(hi/lo): (N,K) bf16 row-major.
// Block = 256 threads = 8 waves. Block tile 128(M) x 64(N).
// B slab (64 rows x 256 K, hi+lo = 64 KB) staged in LDS once per block using
// GLOBAL_LOAD_ASYNC_TO_LDS_B128 (ASYNCcnt) when available; fragments then come
// from ds_load, A streams from global with prefetch.
// Fragment layout per ISA 7.12.2 (16-bit A 16x32, wave32).
// ---------------------------------------------------------------------------
__device__ __forceinline__ v16bf ld_frag(const __bf16* p) {
    v8bf lo = *(const v8bf*)(p);        // K = khalf .. khalf+7
    v8bf hi = *(const v8bf*)(p + 16);   // K = khalf+16 .. khalf+23
    return __builtin_shufflevector(lo, hi, 0, 1, 2, 3, 4, 5, 6, 7,
                                           8, 9, 10, 11, 12, 13, 14, 15);
}

__global__ void __launch_bounds__(256)
gemm_bf16x3_wmma(const __bf16* __restrict__ Ahi, const __bf16* __restrict__ Alo,
                 const __bf16* __restrict__ Bhi, const __bf16* __restrict__ Blo,
                 const float*  __restrict__ bias,
                 float*        __restrict__ C,
                 int lda, int ldb, int ldc) {
    __shared__ __bf16 sBhi[64 * KDIM];   // 32 KB
    __shared__ __bf16 sBlo[64 * KDIM];   // 32 KB

    const int lane  = threadIdx.x & 31;
    const int wid   = threadIdx.x >> 5;
    const int m0    = blockIdx.x * 128 + wid * 16;
    const int n0    = blockIdx.y * 64;
    const int l16   = lane & 15;
    const int khalf = (lane >> 4) * 8;   // 0 or 8: this lane's K-half

    // ---- stage B slab (contiguous since ldb == KDIM): 2 x 32 KB ----
#if defined(HAVE_ASYNC_LDS)
    {
        const char* ghi = (const char*)(Bhi + (size_t)n0 * ldb);
        const char* glo = (const char*)(Blo + (size_t)n0 * ldb);
        char* lhi = (char*)sBhi;
        char* llo = (char*)sBlo;
        for (int i = threadIdx.x * 16; i < 64 * KDIM * 2; i += 256 * 16) {
            __builtin_amdgcn_global_load_async_to_lds_b128(
                (__attribute__((address_space(1))) vint4*)(ghi + i),
                (__attribute__((address_space(3))) vint4*)(lhi + i), 0, 0);
            __builtin_amdgcn_global_load_async_to_lds_b128(
                (__attribute__((address_space(1))) vint4*)(glo + i),
                (__attribute__((address_space(3))) vint4*)(llo + i), 0, 0);
        }
#if __has_builtin(__builtin_amdgcn_s_wait_asynccnt)
        __builtin_amdgcn_s_wait_asynccnt(0);
#else
        asm volatile("s_wait_asynccnt 0x0" ::: "memory");
#endif
    }
#else
    {
        const uint4* shi = (const uint4*)(Bhi + (size_t)n0 * ldb);
        const uint4* slo = (const uint4*)(Blo + (size_t)n0 * ldb);
        uint4* dhi = (uint4*)sBhi;
        uint4* dlo = (uint4*)sBlo;
        for (int i = threadIdx.x; i < (64 * KDIM) / 8; i += 256) {
            dhi[i] = shi[i];
            dlo[i] = slo[i];
        }
    }
#endif
    __syncthreads();

    const __bf16* arow_hi = Ahi + (size_t)(m0 + l16) * lda + khalf;
    const __bf16* arow_lo = Alo + (size_t)(m0 + l16) * lda + khalf;

    v8f acc[4];
#pragma unroll
    for (int t = 0; t < 4; ++t) acc[t] = (v8f){0, 0, 0, 0, 0, 0, 0, 0};

#pragma unroll
    for (int k0 = 0; k0 < KDIM; k0 += 32) {
        __builtin_prefetch(arow_hi + k0 + 64, 0, 3);   // global_prefetch_b8
        __builtin_prefetch(arow_lo + k0 + 64, 0, 3);
        v16bf ahi = ld_frag(arow_hi + k0);
        v16bf alo = ld_frag(arow_lo + k0);
#pragma unroll
        for (int t = 0; t < 4; ++t) {
            const int boff = (t * 16 + l16) * KDIM + k0 + khalf;
            v16bf bhi = ld_frag(&sBhi[boff]);
            v16bf blo = ld_frag(&sBlo[boff]);
            acc[t] = __builtin_amdgcn_wmma_f32_16x16x32_bf16(
                false, ahi, false, bhi, (short)0, acc[t], false, false);
            acc[t] = __builtin_amdgcn_wmma_f32_16x16x32_bf16(
                false, alo, false, bhi, (short)0, acc[t], false, false);
            acc[t] = __builtin_amdgcn_wmma_f32_16x16x32_bf16(
                false, ahi, false, blo, (short)0, acc[t], false, false);
        }
    }

    // Epilogue: D layout = VGPR r: lanes 0-15 -> M=r, lanes 16-31 -> M=r+8
    const int mhalf = (lane >> 4) * 8;
#pragma unroll
    for (int t = 0; t < 4; ++t) {
        const int n  = n0 + t * 16 + l16;
        const float bv = bias[n];
        float* cp = C + (size_t)(m0 + mhalf) * ldc + n;
#pragma unroll
        for (int r = 0; r < 8; ++r)
            cp[(size_t)r * ldc] = acc[t][r] + bv;
    }
}

// ---------------------------------------------------------------------------
// DCNv4 core: one wave per (n, h, w, g); lane = channel within group.
// proj row layout (stride 512 fp32): value[0:256) offset[256:400) mask[400:472)
// Sampling in UNPADDED pixel coords: ix = w + kx + off_x (pad ring is zero,
// so bounds-check against [0,56) is exact). Emits hi/lo bf16 for next GEMM.
// ---------------------------------------------------------------------------
__global__ void dcn_core(const float* __restrict__ proj,
                         __bf16* __restrict__ outhi,
                         __bf16* __restrict__ outlo) {
    const int wave = blockIdx.x * 8 + (threadIdx.x >> 5);   // 0 .. 200703
    const int lane = threadIdx.x & 31;

    const int g  = wave & 7;
    const int sp = wave >> 3;            // n*L + h*W + w
    const int w  = sp % WW;
    const int h  = (sp / WW) % HH;
    const int n  = sp / LL;

    const float* rowp = proj + (size_t)sp * NCAT;
    const float* offp = rowp + 256 + g * (2 * PP);
    const float* mskp = rowp + 400 + g * PP;
    const float* vbase = proj + (size_t)n * LL * NCAT + g * GC + lane;

    float acc = 0.0f;
#pragma unroll
    for (int p = 0; p < PP; ++p) {
        const int kx = p / 3 - 1;        // torch order: p = kx*3 + ky
        const int ky = p % 3 - 1;
        const float ix = (float)(w + kx) + offp[2 * p];
        const float iy = (float)(h + ky) + offp[2 * p + 1];
        const float mk = mskp[p];

        const float x0f = floorf(ix), y0f = floorf(iy);
        const float wx = ix - x0f, wy = iy - y0f;
        const int x0 = (int)x0f, y0 = (int)y0f;
        const int x1 = x0 + 1,   y1 = y0 + 1;

        const bool vx0 = (x0 >= 0) & (x0 < WW);
        const bool vx1 = (x1 >= 0) & (x1 < WW);
        const bool vy0 = (y0 >= 0) & (y0 < HH);
        const bool vy1 = (y1 >= 0) & (y1 < HH);

        const float v00 = (vy0 & vx0) ? vbase[(size_t)(y0 * WW + x0) * NCAT] : 0.0f;
        const float v01 = (vy0 & vx1) ? vbase[(size_t)(y0 * WW + x1) * NCAT] : 0.0f;
        const float v10 = (vy1 & vx0) ? vbase[(size_t)(y1 * WW + x0) * NCAT] : 0.0f;
        const float v11 = (vy1 & vx1) ? vbase[(size_t)(y1 * WW + x1) * NCAT] : 0.0f;

        acc += mk * ((1.0f - wy) * ((1.0f - wx) * v00 + wx * v01)
                   +         wy  * ((1.0f - wx) * v10 + wx * v11));
    }

    const size_t oi = (size_t)sp * CC + g * GC + lane;
    __bf16 h16, l16;
    split_bf16(acc, &h16, &l16);
    outhi[oi] = h16;
    outlo[oi] = l16;
}

// ---------------------------------------------------------------------------
// Launch
// ---------------------------------------------------------------------------
extern "C" void kernel_launch(void* const* d_in, const int* in_sizes, int n_in,
                              void* d_out, int out_size, void* d_ws, size_t ws_size,
                              hipStream_t stream) {
    const float* x     = (const float*)d_in[0];
    const float* vw    = (const float*)d_in[1];
    const float* vb    = (const float*)d_in[2];
    const float* ow    = (const float*)d_in[3];
    const float* ob    = (const float*)d_in[4];
    const float* mw    = (const float*)d_in[5];
    const float* mb    = (const float*)d_in[6];
    const float* outw  = (const float*)d_in[7];
    const float* outb  = (const float*)d_in[8];

    // Workspace carve-up (256B aligned)
    char* ws = (char*)d_ws;
    size_t off = 0;
    auto take = [&](size_t bytes) { char* p = ws + off; off = (off + bytes + 255) & ~(size_t)255; return p; };

    __bf16* xhi     = (__bf16*)take((size_t)MTOT * CC * sizeof(__bf16));     // 12.8 MB
    __bf16* xlo     = (__bf16*)take((size_t)MTOT * CC * sizeof(__bf16));     // 12.8 MB
    __bf16* wchi    = (__bf16*)take((size_t)NCAT * KDIM * sizeof(__bf16));   // 256 KB
    __bf16* wclo    = (__bf16*)take((size_t)NCAT * KDIM * sizeof(__bf16));   // 256 KB
    __bf16* wohi    = (__bf16*)take((size_t)CC * KDIM * sizeof(__bf16));     // 128 KB
    __bf16* wolo    = (__bf16*)take((size_t)CC * KDIM * sizeof(__bf16));     // 128 KB
    float*  biascat = (float*) take((size_t)NCAT * sizeof(float));           // 2 KB
    float*  proj    = (float*) take((size_t)MTOT * NCAT * sizeof(float));    // 51.4 MB
    __bf16* dchi    = (__bf16*)take((size_t)MTOT * CC * sizeof(__bf16));     // 12.8 MB
    __bf16* dclo    = (__bf16*)take((size_t)MTOT * CC * sizeof(__bf16));     // 12.8 MB

    // 1) convert / split / pack
    dcn_convert<<<4096, 256, 0, stream>>>(x, vw, vb, ow, ob, mw, mb, outw,
                                          xhi, xlo, wchi, wclo, wohi, wolo, biascat);

    // 2) fused projection GEMM: (25088 x 256) @ (512 x 256)^T -> proj fp32
    gemm_bf16x3_wmma<<<dim3(MTOT / 128, NCAT / 64), 256, 0, stream>>>(
        xhi, xlo, wchi, wclo, biascat, proj, KDIM, KDIM, NCAT);

    // 3) deformable sampling core -> hi/lo bf16
    dcn_core<<<(NB * LL * GG) / 8, 256, 0, stream>>>(proj, dchi, dclo);

    // 4) output projection: (25088 x 256) @ (256 x 256)^T -> d_out fp32
    gemm_bf16x3_wmma<<<dim3(MTOT / 128, CC / 64), 256, 0, stream>>>(
        dchi, dclo, wohi, wolo, outb, (float*)d_out, KDIM, KDIM, CC);
}
